// SlidingRNN_v2_27676769255801
// MI455X (gfx1250) — matrix operand order repaired and checked by reading, not compiled, not run on hardware
//
#include <hip/hip_runtime.h>
#include <hip/hip_bf16.h>

typedef __bf16 bf16_t;
typedef __bf16 v16bf __attribute__((ext_vector_type(16)));
typedef __bf16 v8bf  __attribute__((ext_vector_type(8)));
typedef float  v8f   __attribute__((ext_vector_type(8)));

#define DEVINL __device__ __forceinline__

static constexpr int B_   = 1024;
static constexpr int T_   = 64;
static constexpr int C_   = 66;
static constexpr int CP_  = 96;    // C padded to multiple of 32 for K-loops
static constexpr int H_   = 1024;
static constexpr int S_   = 12;
static constexpr int L_   = 12;
static constexpr int STW_ = 10;
static constexpr int NF_  = STW_ + T_;   // 74 rolling frames
static constexpr int NOUTP_ = 80;        // out_W rows padded 66 -> 80

static constexpr int RSH_ = H_ + 16;     // LDS row stride (Wh / outW slabs), 32B pad
static constexpr int RSE_ = CP_ + 16;    // LDS row stride (encW slab)

DEVINL v8f wmma_bf16(v16bf a, v16bf b, v8f c) {
  return __builtin_amdgcn_wmma_f32_16x16x32_bf16(false, a, false, b, (short)0, c,
                                                 false, false);
}

union ABu { v16bf v; v8bf h[2]; };

// A fragment (16x32 bf16, MxK): lane holds row (mBase + lane&15);
// lanes 0-15 take K {k0..k0+15}, lanes 16-31 take {k0+16..k0+31}
DEVINL v16bf load_a(const bf16_t* __restrict__ rowPtr, int k0, int lane) {
  int hs = lane >> 4;
  return *(const v16bf*)(rowPtr + k0 + hs * 16);
}

// B fragment (32x16 bf16, KxN) from (N,K) row-major global weight.
DEVINL v16bf load_b(const bf16_t* __restrict__ W, int K, int nRow, int k0, int lane) {
  int hs = lane >> 4;
  return *(const v16bf*)(W + (size_t)nRow * K + k0 + hs * 16);
}

// B fragment from an LDS-staged (rows x rowStride) slab.
DEVINL v16bf load_b_sh(const bf16_t* sh, int row, int rowStride, int k0, int lane) {
  int hs = lane >> 4;
  return *(const v16bf*)(sh + (size_t)row * rowStride + k0 + hs * 16);
}

// CDNA5 async copy: 16B global -> LDS, tracked by ASYNCcnt.
DEVINL void async_copy_b128(const bf16_t* lds_dst, const bf16_t* g_src) {
  unsigned ldsOff = (unsigned)(uintptr_t)lds_dst;  // low 32 bits = LDS offset
  asm volatile("global_load_async_to_lds_b128 %0, %1, off"
               :: "v"(ldsOff), "v"(g_src) : "memory");
}
DEVINL void wait_async0() {
  asm volatile("s_wait_asynccnt 0" ::: "memory");
}

DEVINL float sigmoidf_(float x) { return 1.f / (1.f + __expf(-x)); }

// ---------------------------------------------------------------------------
// Fused GRU step:  h' = (1-z)*tanh(in + r*hn) + z*h
// Block = 128 threads (4 waves), block tile 128(M) x 16(N within H).
// Wh N-slab (3 gates x 16 rows x 1024) staged to LDS via async copies; each
// wave computes M=32 (two 16x16 subtiles, 8 f32 accumulators).
// grid (8, 64).
// ---------------------------------------------------------------------------
__global__ __launch_bounds__(128) void k_gru(
    const bf16_t* __restrict__ h_bf, const float* __restrict__ h_f,
    const bf16_t* __restrict__ aux, long auxStride,
    const bf16_t* __restrict__ Wh, const bf16_t* __restrict__ Wi,
    const float* __restrict__ bi, const float* __restrict__ bh,
    float* __restrict__ ho_f, bf16_t* __restrict__ ho_bf,
    float* __restrict__ wh_out)
{
  __shared__ bf16_t shB[3 * 16 * RSH_];   // ~97.5 KB

  const int tid   = threadIdx.x;
  const int lane  = tid & 31;
  const int wv    = tid >> 5;
  const int nBase = blockIdx.y * 16;
  const int lr    = lane & 15;

  // ---- stage Wh slab: 48 rows x 2048B = 6144 x 16B chunks, 48 per thread ----
#pragma unroll 4
  for (int it = 0; it < 48; ++it) {
    int c   = it * 128 + tid;
    int r   = c >> 7;          // slab row 0..47
    int off = c & 127;         // 16B chunk within row
    int srcRow = (r >> 4) * H_ + nBase + (r & 15);   // gate*H + nBase + rr
    async_copy_b128(shB + (size_t)r * RSH_ + off * 8,
                    Wh + (size_t)srcRow * H_ + off * 8);
  }
  wait_async0();
  __syncthreads();

  const int mBase = blockIdx.x * 128 + wv * 32;

  v8f accR[2] = {}, accZ[2] = {}, accIN[2] = {}, accHN[2] = {};

  const bf16_t* aRow0 = h_bf + (size_t)(mBase + lr) * H_;
  const bf16_t* aRow1 = h_bf + (size_t)(mBase + 16 + lr) * H_;
#pragma unroll 2
  for (int k0 = 0; k0 < H_; k0 += 32) {
    v16bf a0 = load_a(aRow0, k0, lane);
    v16bf a1 = load_a(aRow1, k0, lane);
    v16bf br = load_b_sh(shB, 0 * 16 + lr, RSH_, k0, lane);
    v16bf bz = load_b_sh(shB, 1 * 16 + lr, RSH_, k0, lane);
    v16bf bn = load_b_sh(shB, 2 * 16 + lr, RSH_, k0, lane);
    accR[0]  = wmma_bf16(a0, br, accR[0]);
    accR[1]  = wmma_bf16(a1, br, accR[1]);
    accZ[0]  = wmma_bf16(a0, bz, accZ[0]);
    accZ[1]  = wmma_bf16(a1, bz, accZ[1]);
    accHN[0] = wmma_bf16(a0, bn, accHN[0]);
    accHN[1] = wmma_bf16(a1, bn, accHN[1]);
  }
  const bf16_t* xRow0 = aux + (size_t)(mBase + lr) * (size_t)auxStride;
  const bf16_t* xRow1 = aux + (size_t)(mBase + 16 + lr) * (size_t)auxStride;
#pragma unroll
  for (int k0 = 0; k0 < CP_; k0 += 32) {
    v16bf a0 = load_a(xRow0, k0, lane);
    v16bf a1 = load_a(xRow1, k0, lane);
    v16bf br = load_b(Wi, CP_, 0 * H_ + nBase + lr, k0, lane);
    v16bf bz = load_b(Wi, CP_, 1 * H_ + nBase + lr, k0, lane);
    v16bf bn = load_b(Wi, CP_, 2 * H_ + nBase + lr, k0, lane);
    accR[0]  = wmma_bf16(a0, br, accR[0]);
    accR[1]  = wmma_bf16(a1, br, accR[1]);
    accZ[0]  = wmma_bf16(a0, bz, accZ[0]);
    accZ[1]  = wmma_bf16(a1, bz, accZ[1]);
    accIN[0] = wmma_bf16(a0, bn, accIN[0]);
    accIN[1] = wmma_bf16(a1, bn, accIN[1]);
  }
  const int   col = nBase + lr;
  const float biR = bi[col],          bhR = bh[col];
  const float biZ = bi[H_ + col],     bhZ = bh[H_ + col];
  const float biN = bi[2 * H_ + col], bhN = bh[2 * H_ + col];
  const int   hs  = lane >> 4;
#pragma unroll
  for (int sub = 0; sub < 2; ++sub) {
#pragma unroll
    for (int j = 0; j < 8; ++j) {
      int    row = mBase + sub * 16 + hs * 8 + j;
      size_t idx = (size_t)row * H_ + col;
      float r  = sigmoidf_(accR[sub][j] + biR + bhR);
      float z  = sigmoidf_(accZ[sub][j] + biZ + bhZ);
      float nn = tanhf(accIN[sub][j] + biN + r * (accHN[sub][j] + bhN));
      float hp = h_f[idx];
      float hv = (1.f - z) * nn + z * hp;
      ho_f[idx]  = hv;
      ho_bf[idx] = (bf16_t)hv;
      if (wh_out) wh_out[idx] = hv;
    }
  }
}

// ---------------------------------------------------------------------------
// wst = buf @ enc_W.T : rows (b,s) over B*STW = 10240, K = 96 (padded 66)
// encW N-slab (16 x 96) async-staged to LDS, shared by the 4 waves.
// grid (160, 64), block 128.
// ---------------------------------------------------------------------------
__global__ __launch_bounds__(128) void k_wst(
    const bf16_t* __restrict__ framesT,   // frames + t*CP_
    const bf16_t* __restrict__ encW, const float* __restrict__ encb,
    float* __restrict__ wst)
{
  __shared__ bf16_t shE[16 * RSE_];

  const int tid   = threadIdx.x;
  const int lane  = tid & 31;
  const int wv    = tid >> 5;
  const int nBase = blockIdx.y * 16;
  const int lr    = lane & 15;

  // 16 rows x 12 chunks of 16B = 192 chunks
  for (int c = tid; c < 192; c += 128) {
    int r = c / 12, off = c - r * 12;
    async_copy_b128(shE + (size_t)r * RSE_ + off * 8,
                    encW + (size_t)(nBase + r) * CP_ + off * 8);
  }
  wait_async0();
  __syncthreads();

  const int mBase = (blockIdx.x * 4 + wv) * 16;
  const int m = mBase + lr;
  const int b = m / STW_;
  const int s = m - b * STW_;
  const bf16_t* aRow = framesT + ((size_t)b * NF_ + s) * CP_;

  v8f acc = {};
#pragma unroll
  for (int k0 = 0; k0 < CP_; k0 += 32) {
    v16bf a  = load_a(aRow, k0, lane);
    v16bf bb = load_b_sh(shE, lr, RSE_, k0, lane);
    acc = wmma_bf16(a, bb, acc);
  }
  const int   col = nBase + lr;
  const float eb  = encb[col];
  const int   hs  = lane >> 4;
#pragma unroll
  for (int j = 0; j < 8; ++j) {
    int row = mBase + hs * 8 + j;
    int rb  = row / STW_, rs = row - rb * STW_;
    wst[((size_t)rb * STW_ + rs) * H_ + col] = acc[j] + eb;
  }
}

// ---------------------------------------------------------------------------
// 12-layer mini-MLP + LayerNorm over H, then fold merge_W: g[b,h]=sum_s mW[s]*f[h,s]
// one block per batch element, 1024 threads (one per h), 32 waves.
// ---------------------------------------------------------------------------
__global__ __launch_bounds__(1024) void k_mlp(
    const float* __restrict__ whb, const float* __restrict__ wst,
    const float* __restrict__ hcur,
    const float* __restrict__ seqW, const float* __restrict__ seqb,
    const float* __restrict__ lnA,  const float* __restrict__ lnB,
    const float* __restrict__ mW,   bf16_t* __restrict__ g_bf)
{
  __shared__ float sW[L_ * S_ * S_];
  __shared__ float sB[L_ * S_];
  __shared__ float sMW[S_];
  __shared__ float red[32][2 * S_];
  __shared__ float fin[2 * S_];

  const int tid = threadIdx.x;
  const int b   = blockIdx.x;
  for (int i = tid; i < L_ * S_ * S_; i += 1024) sW[i] = seqW[i];
  for (int i = tid; i < L_ * S_; i += 1024) sB[i] = seqb[i];
  if (tid < S_) sMW[tid] = mW[tid];
  __syncthreads();

  const int h    = tid;
  const int lane = tid & 31, wv = tid >> 5;

  float f[S_];
  f[0] = whb[(size_t)b * H_ + h];
#pragma unroll
  for (int s = 0; s < STW_; ++s) f[1 + s] = wst[((size_t)b * STW_ + s) * H_ + h];
  f[S_ - 1] = hcur[(size_t)b * H_ + h];

  for (int l = 0; l < L_; ++l) {
    float y[S_];
#pragma unroll
    for (int s = 0; s < S_; ++s) {
      float acc = sB[l * S_ + s];
#pragma unroll
      for (int k = 0; k < S_; ++k) acc += f[k] * sW[(l * S_ + s) * S_ + k];
      y[s] = acc;
    }
#pragma unroll
    for (int s = 0; s < S_; ++s) {
      float v = y[s], v2 = v * v;
      for (int off = 16; off > 0; off >>= 1) {
        v  += __shfl_xor(v, off, 32);
        v2 += __shfl_xor(v2, off, 32);
      }
      if (lane == 0) { red[wv][s] = v; red[wv][S_ + s] = v2; }
    }
    __syncthreads();
    if (tid < 2 * S_) {
      float a = 0.f;
      for (int ww = 0; ww < 32; ++ww) a += red[ww][tid];
      fin[tid] = a;
    }
    __syncthreads();
    const float la = lnA[(size_t)l * H_ + h], lb = lnB[(size_t)l * H_ + h];
#pragma unroll
    for (int s = 0; s < S_; ++s) {
      float mu  = fin[s] * (1.f / H_);
      float var = fin[S_ + s] * (1.f / H_) - mu * mu;
      float inv = rsqrtf(var + 1e-5f);
      f[s] += (y[s] - mu) * inv * la + lb;
    }
    __syncthreads();
  }
  float g = 0.f;
#pragma unroll
  for (int s = 0; s < S_; ++s) g += sMW[s] * f[s];
  g_bf[(size_t)b * H_ + h] = (bf16_t)g;
}

// ---------------------------------------------------------------------------
// newf = g @ out_W.T + SW*out_b + merge_b + last ; writes d_out and frames.
// outW N-slab (16 x 1024) async-staged to LDS, shared by the 4 waves.
// grid (16, 5), block 128.
// ---------------------------------------------------------------------------
__global__ __launch_bounds__(128) void k_out(
    const bf16_t* __restrict__ g_bf, const bf16_t* __restrict__ outW,
    const float* __restrict__ outb, const float* __restrict__ mW,
    const float* __restrict__ mb, const float* __restrict__ x,
    float* __restrict__ out, bf16_t* __restrict__ frames, int t)
{
  __shared__ bf16_t shO[16 * RSH_];   // ~32.5 KB

  const int tid   = threadIdx.x;
  const int lane  = tid & 31;
  const int wv    = tid >> 5;
  const int nBase = blockIdx.y * 16;
  const int lr    = lane & 15;

  // 16 rows x 128 chunks of 16B = 2048 chunks, 16 per thread
#pragma unroll 4
  for (int it = 0; it < 16; ++it) {
    int c   = it * 128 + tid;
    int r   = c >> 7;
    int off = c & 127;
    async_copy_b128(shO + (size_t)r * RSH_ + off * 8,
                    outW + (size_t)(nBase + r) * H_ + off * 8);
  }
  wait_async0();
  __syncthreads();

  const int mBase = (blockIdx.x * 4 + wv) * 16;
  const bf16_t* aRow = g_bf + (size_t)(mBase + lr) * H_;
  v8f acc = {};
#pragma unroll 4
  for (int k0 = 0; k0 < H_; k0 += 32) {
    v16bf a  = load_a(aRow, k0, lane);
    v16bf bb = load_b_sh(shO, lr, RSH_, k0, lane);
    acc = wmma_bf16(a, bb, acc);
  }
  float SW = 0.f;
#pragma unroll
  for (int s = 0; s < S_; ++s) SW += mW[s];
  const float mbv = mb[0];
  const int   c   = nBase + lr;
  const int   hs  = lane >> 4;
  if (c < C_) {
    const float ob = outb[c];
#pragma unroll
    for (int j = 0; j < 8; ++j) {
      int   b    = mBase + hs * 8 + j;
      float last = (t == 0) ? x[((size_t)b * T_ + (T_ - 1)) * C_ + c]
                            : out[((size_t)b * T_ + (t - 1)) * C_ + c];
      float v = acc[j] + SW * ob + mbv + last;
      out[((size_t)b * T_ + t) * C_ + c]              = v;
      frames[((size_t)b * NF_ + STW_ + t) * CP_ + c]  = (bf16_t)v;
    }
  }
}

// ------------------------- setup / conversion kernels ----------------------
__global__ void k_pad_bf16(const float* __restrict__ src, int Rs, int Cs,
                           bf16_t* __restrict__ dst, int Rd, int Cd)
{
  size_t idx = (size_t)blockIdx.x * blockDim.x + threadIdx.x;
  if (idx >= (size_t)Rd * Cd) return;
  int r = (int)(idx / Cd), c = (int)(idx % Cd);
  float v = (r < Rs && c < Cs) ? src[(size_t)r * Cs + c] : 0.f;
  dst[idx] = (bf16_t)v;
}

__global__ void k_init_frames(const float* __restrict__ x, bf16_t* __restrict__ frames)
{
  size_t idx = (size_t)blockIdx.x * blockDim.x + threadIdx.x;
  if (idx >= (size_t)B_ * NF_ * CP_) return;
  int    c = (int)(idx % CP_);
  size_t r = idx / CP_;
  int    j = (int)(r % NF_);
  size_t b = r / NF_;
  float  v = 0.f;
  if (j < STW_ && c < C_) v = x[(b * T_ + (T_ - STW_) + j) * C_ + c];
  frames[idx] = (bf16_t)v;
}

__global__ void k_zero_h(float* hf, bf16_t* hbf)
{
  size_t i = (size_t)blockIdx.x * blockDim.x + threadIdx.x;
  if (i < (size_t)B_ * H_) { hf[i] = 0.f; hbf[i] = (bf16_t)0.f; }
}

// ---------------------------------------------------------------------------
extern "C" void kernel_launch(void* const* d_in, const int* in_sizes, int n_in,
                              void* d_out, int out_size, void* d_ws, size_t ws_size,
                              hipStream_t stream)
{
  (void)in_sizes; (void)n_in; (void)out_size; (void)ws_size;
  const float* x    = (const float*)d_in[0];
  const float* Wi   = (const float*)d_in[1];
  const float* Wh   = (const float*)d_in[2];
  const float* bi   = (const float*)d_in[3];
  const float* bh   = (const float*)d_in[4];
  const float* encW = (const float*)d_in[5];
  const float* encb = (const float*)d_in[6];
  const float* seqW = (const float*)d_in[7];
  const float* seqb = (const float*)d_in[8];
  const float* lnA  = (const float*)d_in[9];
  const float* lnB  = (const float*)d_in[10];
  const float* outW = (const float*)d_in[11];
  const float* outb = (const float*)d_in[12];
  const float* mW   = (const float*)d_in[13];
  const float* mb   = (const float*)d_in[14];
  float* out = (float*)d_out;

  char* p = (char*)d_ws;
  auto alloc = [&](size_t bytes) {
    char* r = p; p += (bytes + 255) & ~(size_t)255; return r;
  };
  bf16_t* Wh_bf   = (bf16_t*)alloc((size_t)3 * H_ * H_ * 2);
  bf16_t* Wi_bf   = (bf16_t*)alloc((size_t)3 * H_ * CP_ * 2);
  bf16_t* encW_bf = (bf16_t*)alloc((size_t)H_ * CP_ * 2);
  bf16_t* outW_bf = (bf16_t*)alloc((size_t)NOUTP_ * H_ * 2);
  bf16_t* xpad    = (bf16_t*)alloc((size_t)B_ * T_ * CP_ * 2);
  bf16_t* frames  = (bf16_t*)alloc((size_t)B_ * NF_ * CP_ * 2);
  float*  hf[2];  bf16_t* hbf[2];
  hf[0]  = (float*)alloc((size_t)B_ * H_ * 4);
  hf[1]  = (float*)alloc((size_t)B_ * H_ * 4);
  hbf[0] = (bf16_t*)alloc((size_t)B_ * H_ * 2);
  hbf[1] = (bf16_t*)alloc((size_t)B_ * H_ * 2);
  float*  whb  = (float*)alloc((size_t)B_ * H_ * 4);
  float*  wst  = (float*)alloc((size_t)B_ * STW_ * H_ * 4);
  bf16_t* g_bf = (bf16_t*)alloc((size_t)B_ * H_ * 2);

  auto g1 = [](size_t n) { return dim3((unsigned)((n + 255) / 256)); };

  // one-time (per launch) conversions — deterministic, replay-safe
  k_pad_bf16<<<g1((size_t)3*H_*H_), 256, 0, stream>>>(Wh, 3*H_, H_, Wh_bf, 3*H_, H_);
  k_pad_bf16<<<g1((size_t)3*H_*CP_), 256, 0, stream>>>(Wi, 3*H_, C_, Wi_bf, 3*H_, CP_);
  k_pad_bf16<<<g1((size_t)H_*CP_), 256, 0, stream>>>(encW, H_, C_, encW_bf, H_, CP_);
  k_pad_bf16<<<g1((size_t)NOUTP_*H_), 256, 0, stream>>>(outW, C_, H_, outW_bf, NOUTP_, H_);
  k_pad_bf16<<<g1((size_t)B_*T_*CP_), 256, 0, stream>>>(x, B_*T_, C_, xpad, B_*T_, CP_);
  k_init_frames<<<g1((size_t)B_*NF_*CP_), 256, 0, stream>>>(x, frames);
  k_zero_h<<<g1((size_t)B_*H_), 256, 0, stream>>>(hf[0], hbf[0]);

  const dim3 gruGrid(8, 64), blk(128);
  int pc = 0;

  // encoder: 63 sequential GRU steps over x[:, :-1]
  for (int t = 0; t < T_ - 1; ++t) {
    k_gru<<<gruGrid, blk, 0, stream>>>(hbf[pc], hf[pc],
        xpad + (size_t)t * CP_, (long)T_ * CP_,
        Wh_bf, Wi_bf, bi, bh, hf[1 - pc], hbf[1 - pc], nullptr);
    pc ^= 1;
  }

  // decoder: 64 steps; step-0 GRU output == window_history (same inputs)
  for (int t = 0; t < T_; ++t) {
    k_gru<<<gruGrid, blk, 0, stream>>>(hbf[pc], hf[pc],
        frames + (size_t)(STW_ - 1 + t) * CP_, (long)NF_ * CP_,
        Wh_bf, Wi_bf, bi, bh, hf[1 - pc], hbf[1 - pc],
        (t == 0) ? whb : nullptr);
    pc ^= 1;
    k_wst<<<dim3(160, 64), blk, 0, stream>>>(frames + (size_t)t * CP_,
                                             encW_bf, encb, wst);
    k_mlp<<<dim3(B_), dim3(1024), 0, stream>>>(whb, wst, hf[pc],
                                               seqW, seqb, lnA, lnB, mW, g_bf);
    k_out<<<dim3(16, 5), blk, 0, stream>>>(g_bf, outW_bf, outb, mW, mb,
                                           x, out, frames, t);
  }
}
